// AbsoluteRelativePositionEmbedding_6253472383724
// MI455X (gfx1250) — matrix operand order repaired and checked by reading, not compiled, not run on hardware
//
#include <hip/hip_runtime.h>

// ---------------------------------------------------------------------------
// DGCNN-style point network on MI455X (gfx1250, wave32, WMMA).
// All 1x1 convs run as bf16 WMMA GEMMs (v_wmma_f32_16x16x32_bf16, f32 acc).
// Each wave computes a 32(out-ch) x 64(position) macro-tile: 8 WMMAs per
// K-chunk against 4 A-loads + 8 B-loads (all b128). GroupNorm fused.
// ---------------------------------------------------------------------------

#define BB 8
#define NN 2048
#define KK 16
#define GN 8          // group-norm groups
#define EPSV 1e-5f

typedef __attribute__((ext_vector_type(16))) __bf16 v16bf;
typedef __attribute__((ext_vector_type(8)))  __bf16 v8bf;
typedef __attribute__((ext_vector_type(8)))  float  v8f;

#define WMMA_BF16(A, Bm, C) \
  __builtin_amdgcn_wmma_f32_16x16x32_bf16(false, (A), false, (Bm), (short)0, (C), false, false)

// ------------------------------- utility -----------------------------------
__global__ void zero_f32_kernel(float* __restrict__ p, int n) {
  int i = blockIdx.x * 256 + threadIdx.x;
  if (i < n) p[i] = 0.0f;
}

// Convert f32 weights (Cout x CinReal) to bf16 zero-padded (Cout x CinPad).
__global__ void wprep_kernel(const float* __restrict__ W, __bf16* __restrict__ Wb,
                             int CinReal, int CinPad, int Cout) {
  const int t = blockIdx.x * 256 + threadIdx.x;
  if (t >= Cout * CinPad) return;
  const int o = t / CinPad, c = t % CinPad;
  Wb[t] = (__bf16)((c < CinReal) ? W[(size_t)o * CinReal + c] : 0.0f);
}

// ------------------------------- KNN ---------------------------------------
// grid: BB * (NN/256) blocks, 256 threads. Batch points staged in LDS (24KB).
__global__ __launch_bounds__(256)
void knn_kernel(const float* __restrict__ pts, int* __restrict__ idx_out) {
  __shared__ float sx[NN], sy[NN], sz[NN];
  const int b = blockIdx.x >> 3;            // NN/256 == 8 chunks per batch
  const int n = ((blockIdx.x & 7) << 8) + threadIdx.x;
  for (int t = threadIdx.x; t < NN; t += 256) {
    sx[t] = pts[(size_t)(b * 3 + 0) * NN + t];
    sy[t] = pts[(size_t)(b * 3 + 1) * NN + t];
    sz[t] = pts[(size_t)(b * 3 + 2) * NN + t];
  }
  __syncthreads();
  const float qx = sx[n], qy = sy[n], qz = sz[n];
  float bd[KK + 1];
  int   bi[KK + 1];
#pragma unroll
  for (int t = 0; t <= KK; ++t) { bd[t] = 3.0e38f; bi[t] = 0; }
  for (int j = 0; j < NN; ++j) {
    const float dx = sx[j] - qx, dy = sy[j] - qy, dz = sz[j] - qz;
    float d = dx * dx + dy * dy + dz * dz;
    if (d < bd[KK]) {
      float cd = d; int ci = j;
#pragma unroll
      for (int t = 0; t <= KK; ++t) {          // sorted insert (bubble carry)
        const float od = bd[t]; const int oi = bi[t];
        const bool take = cd < od;
        bd[t] = take ? cd : od;  bi[t] = take ? ci : oi;
        cd    = take ? od : cd;  ci    = take ? oi : ci;
      }
    }
  }
  int* op = idx_out + ((size_t)(b * NN + n)) * KK;
#pragma unroll
  for (int t = 0; t < KK; ++t) op[t] = bi[t + 1];   // drop self
}

// --------------------------- edge features ---------------------------------
// X0: position-major bf16, Cin padded 6 -> 32.  m = (b*NN+n)*KK + k
__global__ __launch_bounds__(256)
void edge_build_kernel(const float* __restrict__ pts, const int* __restrict__ idx,
                       __bf16* __restrict__ X0) {
  const int t = blockIdx.x * 256 + threadIdx.x;
  if (t >= BB * NN * KK) return;
  const int n = (t / KK) % NN;
  const int b = t / (KK * NN);
  const int j = idx[t];
  const float cx = pts[(size_t)(b * 3 + 0) * NN + n];
  const float cy = pts[(size_t)(b * 3 + 1) * NN + n];
  const float cz = pts[(size_t)(b * 3 + 2) * NN + n];
  const float nx = pts[(size_t)(b * 3 + 0) * NN + j];
  const float ny = pts[(size_t)(b * 3 + 1) * NN + j];
  const float nz = pts[(size_t)(b * 3 + 2) * NN + j];
  v8bf lo = {};
  lo[0] = (__bf16)cx; lo[1] = (__bf16)cy; lo[2] = (__bf16)cz;
  lo[3] = (__bf16)(nx - cx); lo[4] = (__bf16)(ny - cy); lo[5] = (__bf16)(nz - cz);
  v8bf z = {};
  __bf16* p = X0 + (size_t)t * 32;
  *(v8bf*)(p + 0)  = lo;
  *(v8bf*)(p + 8)  = z;
  *(v8bf*)(p + 16) = z;
  *(v8bf*)(p + 24) = z;
}

// ------------------------------ WMMA GEMM ----------------------------------
// Y[m*Cout + o] = sum_c Wb[o,c] * X[m*Cin + c] + bias[o]   (bf16 in, f32 acc)
// Wave macro-tile: 32 out-channels (2 A tiles) x 64 positions (4 B tiles)
// -> 8 WMMAs per 32-wide K-chunk. 8 waves / block. Fused GroupNorm stats.
__global__ __launch_bounds__(256)
void gemm_gn_stats_kernel(const __bf16* __restrict__ Wb, const float* __restrict__ bias,
                          const __bf16* __restrict__ X, __bf16* __restrict__ Y,
                          float* __restrict__ ssum, float* __restrict__ ssq,
                          int Cin, int Cout, int Mb) {
  const int lane = threadIdx.x & 31;
  const int wave = threadIdx.x >> 5;
  const int m0 = (blockIdx.x * 8 + wave) * 64;   // 4 position tiles
  const int o0 = blockIdx.y * 32;                // 2 out-channel tiles
  const int h  = lane >> 4;                      // lane half
  const int nl = lane & 15;
  const size_t wrow0 = (size_t)(o0 + nl) * Cin + h * 16;       // A tile 0 row
  const size_t wrow1 = (size_t)(o0 + 16 + nl) * Cin + h * 16;  // A tile 1 row
  const __bf16* xp = X + (size_t)(m0 + nl) * Cin + h * 16;
  const size_t xs = (size_t)16 * Cin;            // position-tile stride

  v8f a00 = {}, a01 = {}, a02 = {}, a03 = {};
  v8f a10 = {}, a11 = {}, a12 = {}, a13 = {};
  for (int c0 = 0; c0 < Cin; c0 += 32) {
    // A: two 16x32 bf16 weight tiles; per-lane two contiguous 16B loads each
    const v8bf w0lo = *(const v8bf*)(Wb + wrow0 + c0);
    const v8bf w0hi = *(const v8bf*)(Wb + wrow0 + c0 + 8);
    const v8bf w1lo = *(const v8bf*)(Wb + wrow1 + c0);
    const v8bf w1hi = *(const v8bf*)(Wb + wrow1 + c0 + 8);
    const v16bf wA0 = __builtin_shufflevector(w0lo, w0hi, 0, 1, 2, 3, 4, 5, 6, 7,
                                              8, 9, 10, 11, 12, 13, 14, 15);
    const v16bf wA1 = __builtin_shufflevector(w1lo, w1hi, 0, 1, 2, 3, 4, 5, 6, 7,
                                              8, 9, 10, 11, 12, 13, 14, 15);
    // B: four 32x16 bf16 activation tiles; 32B contiguous per lane each
    const v16bf b0 = *(const v16bf*)(xp + c0);
    const v16bf b1 = *(const v16bf*)(xp + xs + c0);
    const v16bf b2 = *(const v16bf*)(xp + 2 * xs + c0);
    const v16bf b3 = *(const v16bf*)(xp + 3 * xs + c0);
    a00 = WMMA_BF16(wA0, b0, a00);
    a01 = WMMA_BF16(wA0, b1, a01);
    a02 = WMMA_BF16(wA0, b2, a02);
    a03 = WMMA_BF16(wA0, b3, a03);
    a10 = WMMA_BF16(wA1, b0, a10);
    a11 = WMMA_BF16(wA1, b1, a11);
    a12 = WMMA_BF16(wA1, b2, a12);
    a13 = WMMA_BF16(wA1, b3, a13);
  }

  // D element r -> channel oc + r, position (m0 + 16t + nl).
  v8f* accs[2][4] = { { &a00, &a01, &a02, &a03 }, { &a10, &a11, &a12, &a13 } };
  const int bidx = m0 / Mb;

#pragma unroll
  for (int ot = 0; ot < 2; ++ot) {
    const int oc = o0 + 16 * ot + 8 * h;
    float bv[8], s8[8], q8[8];
#pragma unroll
    for (int r = 0; r < 8; ++r) { bv[r] = bias[oc + r]; s8[r] = 0.0f; q8[r] = 0.0f; }

#pragma unroll
    for (int t = 0; t < 4; ++t) {
      v8f& a4 = *accs[ot][t];
      v8bf yv;
#pragma unroll
      for (int r = 0; r < 8; ++r) {
        const float v = a4[r] + bv[r];
        s8[r] += v;
        q8[r] += v * v;
        yv[r] = (__bf16)v;
      }
      *(v8bf*)(Y + (size_t)(m0 + 16 * t + nl) * Cout + oc) = yv;
    }

    // fused GroupNorm stats: reduce over the 64 positions of the macro-tile
#pragma unroll
    for (int r = 0; r < 8; ++r) {
      float s = s8[r], q = q8[r];
#pragma unroll
      for (int off = 1; off < 16; off <<= 1) {   // stays within lane half
        s += __shfl_xor(s, off, 32);
        q += __shfl_xor(q, off, 32);
      }
      if (nl == 0) {
        atomicAdd(&ssum[(size_t)bidx * Cout + oc + r], s);
        atomicAdd(&ssq [(size_t)bidx * Cout + oc + r], q);
      }
    }
  }
}

// --------------------------- GroupNorm finalize ----------------------------
__global__ void gn_finalize_kernel(const float* __restrict__ ssum,
                                   const float* __restrict__ ssq,
                                   float* __restrict__ mean, float* __restrict__ rsig,
                                   int Cout, int Mb) {
  const int t = threadIdx.x;                    // launch BB*GN = 64 threads
  if (t >= BB * GN) return;
  const int b = t / GN, g = t % GN;
  const int cpg = Cout / GN;
  float s = 0.0f, q = 0.0f;
  for (int c = g * cpg; c < (g + 1) * cpg; ++c) {
    s += ssum[(size_t)b * Cout + c];
    q += ssq [(size_t)b * Cout + c];
  }
  const float cnt = (float)cpg * (float)Mb;
  const float mu  = s / cnt;
  const float var = q / cnt - mu * mu;
  mean[t] = mu;
  rsig[t] = rsqrtf(var + EPSV);
}

// ------------------------ GroupNorm apply + ELU ----------------------------
// FINAL=false: in-place bf16 (feeds next WMMA layer).
// FINAL=true : write f32 to d_out as (B, Cout, N).
template <bool FINAL>
__global__ __launch_bounds__(256)
void gn_apply_elu_kernel(__bf16* __restrict__ Y,
                         const float* __restrict__ mean, const float* __restrict__ rsig,
                         const float* __restrict__ gamma, const float* __restrict__ beta,
                         int Cout, int Mb, int Mtot, float* __restrict__ outF) {
  const int t = blockIdx.x * 256 + threadIdx.x;
  const int perm = Cout / 8;
  const int m  = t / perm;
  const int c0 = (t % perm) * 8;
  if (m >= Mtot) return;
  const int b   = m / Mb;
  const int cpg = Cout / GN;
  const int g   = c0 / cpg;                     // 8-chunk never crosses a group
  const float mu = mean[b * GN + g];
  const float rs = rsig[b * GN + g];
  v8bf yv = *(v8bf*)(Y + (size_t)m * Cout + c0);
  if (FINAL) {
    const int n = m % Mb;                       // Mb == NN for the final stage
#pragma unroll
    for (int r = 0; r < 8; ++r) {
      const int c = c0 + r;
      float x = ((float)yv[r] - mu) * rs * gamma[c] + beta[c];
      x = (x > 0.0f) ? x : expm1f(x);
      outF[((size_t)(b * Cout + c)) * NN + n] = x;
    }
  } else {
#pragma unroll
    for (int r = 0; r < 8; ++r) {
      const int c = c0 + r;
      float x = ((float)yv[r] - mu) * rs * gamma[c] + beta[c];
      x = (x > 0.0f) ? x : expm1f(x);
      yv[r] = (__bf16)x;
    }
    *(v8bf*)(Y + (size_t)m * Cout + c0) = yv;
  }
}

// ------------------------------ max over K ---------------------------------
// X3: ((b*NN+n)*KK + k)*256 + c  ->  X4: (b*NN+n)*256 + c
__global__ __launch_bounds__(256)
void maxk_kernel(const __bf16* __restrict__ X3, __bf16* __restrict__ X4) {
  const int t = blockIdx.x * 256 + threadIdx.x; // per (m2, 8 channels)
  const int m2 = t >> 5;
  const int c0 = (t & 31) * 8;
  if (m2 >= BB * NN) return;
  float best[8];
#pragma unroll
  for (int r = 0; r < 8; ++r) best[r] = -3.0e38f;
  for (int k = 0; k < KK; ++k) {
    const v8bf v = *(const v8bf*)(X3 + ((size_t)m2 * KK + k) * 256 + c0);
#pragma unroll
    for (int r = 0; r < 8; ++r) best[r] = fmaxf(best[r], (float)v[r]);
  }
  v8bf o;
#pragma unroll
  for (int r = 0; r < 8; ++r) o[r] = (__bf16)best[r];
  *(v8bf*)(X4 + (size_t)m2 * 256 + c0) = o;
}

// ------------------------------- host side ---------------------------------
extern "C" void kernel_launch(void* const* d_in, const int* in_sizes, int n_in,
                              void* d_out, int out_size, void* d_ws, size_t ws_size,
                              hipStream_t stream) {
  (void)in_sizes; (void)n_in; (void)out_size; (void)ws_size;
  const float* pts = (const float*)d_in[0];
  struct Layer { const float *w, *b, *g, *be; int cin, cinPad, cout; };
  const Layer L[5] = {
    { (const float*)d_in[1],  (const float*)d_in[2],  (const float*)d_in[3],  (const float*)d_in[4],    6,  32,   64 },
    { (const float*)d_in[5],  (const float*)d_in[6],  (const float*)d_in[7],  (const float*)d_in[8],   64,  64,  128 },
    { (const float*)d_in[9],  (const float*)d_in[10], (const float*)d_in[11], (const float*)d_in[12], 128, 128,  256 },
    { (const float*)d_in[13], (const float*)d_in[14], (const float*)d_in[15], (const float*)d_in[16], 256, 256,  512 },
    { (const float*)d_in[17], (const float*)d_in[18], (const float*)d_in[19], (const float*)d_in[20], 512, 512, 1024 },
  };

  char* ws = (char*)d_ws;
  int*    idxp  = (int*)ws;                                   // 1 MB
  float*  ssum  = (float*)(ws + (1u << 20));                  // 32 KB
  float*  ssq   = ssum + BB * 1024;                           // 32 KB
  float*  meanp = ssq + BB * 1024;                            // 64 f32
  float*  rsigp = meanp + BB * GN;
  __bf16* wbase = (__bf16*)(ws + (1u << 20) + (128u << 10));  // 2 MB region
  __bf16* wb[5];
  {
    size_t off = 0;
    for (int i = 0; i < 5; ++i) {
      wb[i] = wbase + off;
      off += (size_t)L[i].cout * L[i].cinPad;
    }
  }
  __bf16* buf0 = (__bf16*)(ws + (1u << 20) + (128u << 10) + (2u << 20)); // 64 MB
  __bf16* buf1 = (__bf16*)((char*)buf0 + (64u << 20));                   // 128 MB

  const int M1 = BB * NN * KK;   // 262144 positions (stage 1)
  const int M2 = BB * NN;        // 16384 positions (stage 2)
  const int Mb1 = NN * KK, Mb2 = NN;

  // 0) weight conversion to padded bf16
  for (int i = 0; i < 5; ++i) {
    const int ne = L[i].cout * L[i].cinPad;
    wprep_kernel<<<(ne + 255) / 256, 256, 0, stream>>>(L[i].w, wb[i], L[i].cin,
                                                       L[i].cinPad, L[i].cout);
  }

  // 1) KNN + edge features -> X0 in buf0 (32-ch padded bf16)
  knn_kernel<<<BB * (NN / 256), 256, 0, stream>>>(pts, idxp);
  edge_build_kernel<<<(M1 + 255) / 256, 256, 0, stream>>>(pts, idxp, buf0);

  // 2) stage-1 blocks: buf0 -> buf1 -> buf0 -> buf1 (apply in place)
  __bf16* xin  = buf0;
  __bf16* yout = buf1;
  for (int i = 0; i < 3; ++i) {
    zero_f32_kernel<<<(2 * BB * 1024 + 255) / 256, 256, 0, stream>>>(ssum, 2 * BB * 1024);
    dim3 gg(M1 / 512, L[i].cout / 32);
    gemm_gn_stats_kernel<<<gg, 256, 0, stream>>>(wb[i], L[i].b, xin, yout, ssum, ssq,
                                                 L[i].cinPad, L[i].cout, Mb1);
    gn_finalize_kernel<<<1, 64, 0, stream>>>(ssum, ssq, meanp, rsigp, L[i].cout, Mb1);
    const int at = M1 * L[i].cout / 8;
    gn_apply_elu_kernel<false><<<(at + 255) / 256, 256, 0, stream>>>(
        yout, meanp, rsigp, L[i].g, L[i].be, L[i].cout, Mb1, M1, nullptr);
    __bf16* tmp = xin; xin = yout; yout = tmp;
  }
  // after loop: activations (256-ch) live in buf1

  // 3) max over K: buf1 -> buf0 (256-ch, M2 positions)
  maxk_kernel<<<(M2 * 32 + 255) / 256, 256, 0, stream>>>(buf1, buf0);

  // 4) stage-2 blocks
  // L4: buf0 -> buf1
  {
    zero_f32_kernel<<<(2 * BB * 1024 + 255) / 256, 256, 0, stream>>>(ssum, 2 * BB * 1024);
    dim3 gg(M2 / 512, L[3].cout / 32);
    gemm_gn_stats_kernel<<<gg, 256, 0, stream>>>(wb[3], L[3].b, buf0, buf1, ssum, ssq,
                                                 L[3].cinPad, L[3].cout, Mb2);
    gn_finalize_kernel<<<1, 64, 0, stream>>>(ssum, ssq, meanp, rsigp, L[3].cout, Mb2);
    const int at = M2 * L[3].cout / 8;
    gn_apply_elu_kernel<false><<<(at + 255) / 256, 256, 0, stream>>>(
        buf1, meanp, rsigp, L[3].g, L[3].be, L[3].cout, Mb2, M2, nullptr);
  }
  // L5: buf1 -> buf0, final apply writes f32 (B, 1024, N) to d_out
  {
    zero_f32_kernel<<<(2 * BB * 1024 + 255) / 256, 256, 0, stream>>>(ssum, 2 * BB * 1024);
    dim3 gg(M2 / 512, L[4].cout / 32);
    gemm_gn_stats_kernel<<<gg, 256, 0, stream>>>(wb[4], L[4].b, buf1, buf0, ssum, ssq,
                                                 L[4].cinPad, L[4].cout, Mb2);
    gn_finalize_kernel<<<1, 64, 0, stream>>>(ssum, ssq, meanp, rsigp, L[4].cout, Mb2);
    const int at = M2 * L[4].cout / 8;
    gn_apply_elu_kernel<true><<<(at + 255) / 256, 256, 0, stream>>>(
        buf0, meanp, rsigp, L[4].g, L[4].be, L[4].cout, Mb2, M2, (float*)d_out);
  }
}
